// RadZeroLoss_30915174597224
// MI455X (gfx1250) — matrix operand order repaired and checked by reading, not compiled, not run on hardware
//
#include <hip/hip_runtime.h>
#include <hip/hip_bf16.h>
#include <math.h>

// ---------------- problem constants (from setup_inputs) ----------------
constexpr int N_TXT = 1024;   // text rows
constexpr int B_IMG = 64;     // images
constexpr int L_TOK = 197;    // tokens
constexpr int L_PAD = 224;    // tokens padded to multiple of 32 (7 * 32)
constexpr int D_DIM = 768;    // feature dim (= 24 * 32 = 48 * 16)

// ---------------- workspace layout (bytes) ----------------
constexpr size_t OFF_TF16 = 0;                                        // bf16 [N][D]
constexpr size_t OFF_QN   = OFF_TF16 + (size_t)N_TXT * D_DIM * 2;     // f32  [N][D]
constexpr size_t OFF_VT16 = OFF_QN   + (size_t)N_TXT * D_DIM * 4;     // bf16 [B][Lpad][D]
constexpr size_t OFF_VTT  = OFF_VT16 + (size_t)B_IMG * L_PAD * D_DIM * 2; // bf16 [B][D][Lpad]
constexpr size_t OFF_LOG  = OFF_VTT  + (size_t)B_IMG * D_DIM * L_PAD * 2; // f32 [N][B]
constexpr size_t OFF_CS   = OFF_LOG  + (size_t)N_TXT * B_IMG * 4;     // f32 [B] colsum
constexpr size_t OFF_CP   = OFF_CS + 256;                             // f32 [B] colpos

// ---------------- types for WMMA ----------------
typedef __attribute__((ext_vector_type(16))) __bf16 v16bf;
typedef __attribute__((ext_vector_type(8)))  float  v8f;

struct BF16x16 { uint4 lo, hi; };
static __device__ __forceinline__ v16bf mk16(uint4 lo, uint4 hi) {
    BF16x16 t{lo, hi};
    return __builtin_bit_cast(v16bf, t);
}

static __device__ __forceinline__ unsigned short f2bf(float f) {
    unsigned u = __builtin_bit_cast(unsigned, f);
    u += 0x7FFFu + ((u >> 16) & 1u);   // round-to-nearest-even
    return (unsigned short)(u >> 16);
}

static __device__ __forceinline__ float block_reduce(float v, float* buf, int tid, int nthr) {
    buf[tid] = v;
    __syncthreads();
    for (int s = nthr >> 1; s > 0; s >>= 1) {
        if (tid < s) buf[tid] += buf[tid + s];
        __syncthreads();
    }
    float r = buf[0];
    __syncthreads();
    return r;
}

// ================= Kernel A: LayerNorm + L2-norm of text =================
__global__ __launch_bounds__(256)
void ln_text_kernel(const float* __restrict__ x, const float* __restrict__ gamma,
                    const float* __restrict__ beta,
                    unsigned short* __restrict__ tf16, float* __restrict__ qn) {
    __shared__ float buf[256];
    const int n = blockIdx.x, tid = threadIdx.x;
    const float* row = x + (size_t)n * D_DIM;
    float v[3];
    float s = 0.f, ss = 0.f;
    for (int i = 0; i < 3; ++i) {
        v[i] = row[tid + 256 * i];
        s += v[i];
        ss += v[i] * v[i];
    }
    const float S  = block_reduce(s,  buf, tid, 256);
    const float SS = block_reduce(ss, buf, tid, 256);
    const float mean = S * (1.f / D_DIM);
    const float var  = SS * (1.f / D_DIM) - mean * mean;
    const float inv  = rsqrtf(var + 1e-5f);
    float y[3], ysq = 0.f;
    for (int i = 0; i < 3; ++i) {
        int d = tid + 256 * i;
        y[i] = (v[i] - mean) * inv * gamma[d] + beta[d];
        ysq += y[i] * y[i];
    }
    const float SY = block_reduce(ysq, buf, tid, 256);
    const float invn = 1.f / fmaxf(sqrtf(SY), 1e-12f);
    for (int i = 0; i < 3; ++i) {
        int d = tid + 256 * i;
        tf16[(size_t)n * D_DIM + d] = f2bf(y[i]);
        qn  [(size_t)n * D_DIM + d] = y[i] * invn;
    }
}

// ============ Kernel B: LayerNorm of vision (+ transpose, zero-pad) ============
__global__ __launch_bounds__(256)
void ln_vis_kernel(const float* __restrict__ x, const float* __restrict__ gamma,
                   const float* __restrict__ beta,
                   unsigned short* __restrict__ vt16, unsigned short* __restrict__ vtT) {
    __shared__ float buf[256];
    const int l = blockIdx.x, b = blockIdx.y, tid = threadIdx.x;
    unsigned short* orow = vt16 + ((size_t)b * L_PAD + l) * D_DIM;
    if (l >= L_TOK) {  // zero padding rows
        for (int i = 0; i < 3; ++i) {
            int d = tid + 256 * i;
            orow[d] = 0;
            vtT[((size_t)b * D_DIM + d) * L_PAD + l] = 0;
        }
        return;
    }
    const float* row = x + ((size_t)b * L_TOK + l) * D_DIM;
    float v[3];
    float s = 0.f, ss = 0.f;
    for (int i = 0; i < 3; ++i) {
        v[i] = row[tid + 256 * i];
        s += v[i];
        ss += v[i] * v[i];
    }
    const float S  = block_reduce(s,  buf, tid, 256);
    const float SS = block_reduce(ss, buf, tid, 256);
    const float mean = S * (1.f / D_DIM);
    const float var  = SS * (1.f / D_DIM) - mean * mean;
    const float inv  = rsqrtf(var + 1e-5f);
    for (int i = 0; i < 3; ++i) {
        int d = tid + 256 * i;
        unsigned short h = f2bf((v[i] - mean) * inv * gamma[d] + beta[d]);
        orow[d] = h;
        vtT[((size_t)b * D_DIM + d) * L_PAD + l] = h;
    }
}

// ====== Kernel C: fused attention + aggregation + normalized dot (per b, 16-row tile) ======
// grid (N/16, B), one wave32 per block. WMMA bf16 16x16x32.
__global__ __launch_bounds__(32)
void attn_kernel(const unsigned short* __restrict__ tf16, const float* __restrict__ qn,
                 const unsigned short* __restrict__ vt16, const unsigned short* __restrict__ vtT,
                 float* __restrict__ logits) {
    constexpr int NLT = L_PAD / 16;   // 14 token tiles of 16 (but scores need only 13 for L<=208)
    constexpr int NLT_S = 13;         // 13*16 = 208 >= 197
    __shared__ float          sc[16][NLT_S * 16];    // raw scores
    __shared__ unsigned short at[16][L_PAD];         // exp(s - max), bf16, zero-padded

    const int ntile = blockIdx.x, b = blockIdx.y;
    const int lane = threadIdx.x;
    const int col  = lane & 15;
    const int hi   = lane >> 4;

    // ---------- Phase 1: scores = tf16_tile [16xD] * vt16[b]^T ----------
    const unsigned short* arow = tf16 + (size_t)(ntile * 16 + col) * D_DIM;  // A row m=col
    for (int lt = 0; lt < NLT_S; ++lt) {
        const unsigned short* brow = vt16 + ((size_t)b * L_PAD + lt * 16 + col) * D_DIM; // B col n=col -> token
        v8f c; for (int r = 0; r < 8; ++r) c[r] = 0.f;
        for (int ks = 0; ks < D_DIM / 32; ++ks) {
            const int ka = ks * 32 + 8 * hi;
            uint4 a0 = *(const uint4*)(arow + ka);
            uint4 a1 = *(const uint4*)(arow + ka + 16);
            const int kb = ks * 32 + 8 * hi;
            uint4 b0 = *(const uint4*)(brow + kb);
            uint4 b1 = *(const uint4*)(brow + kb + 16);
            c = __builtin_amdgcn_wmma_f32_16x16x32_bf16(
                    false, mk16(a0, a1), false, mk16(b0, b1), (short)0, c, false, false);
        }
        const float scale = 0.03608439182435161f;  // 1/sqrt(768)
        for (int r = 0; r < 8; ++r) sc[r + 8 * hi][lt * 16 + col] = c[r] * scale;
    }
    __syncthreads();

    // ---------- Phase 2: row softmax numerator only (denominator cancels in l2norm) ----------
    {
        const int row = lane >> 1, part = lane & 1;
        float mx = -1e30f;
        for (int j = part; j < L_TOK; j += 2) mx = fmaxf(mx, sc[row][j]);
        mx = fmaxf(mx, __shfl_xor(mx, 1, 32));
        for (int j = part; j < L_TOK; j += 2) at[row][j] = f2bf(expf(sc[row][j] - mx));
        for (int j = L_TOK + part; j < L_PAD; j += 2) at[row][j] = 0;
    }
    __syncthreads();

    // ---------- Phase 3: aggregated = p @ vt ; fold into sumsq & dot(qn) ----------
    float accS[8], accD[8];
    for (int r = 0; r < 8; ++r) { accS[r] = 0.f; accD[r] = 0.f; }
    const float* qbase = qn + (size_t)(ntile * 16) * D_DIM;
    const unsigned short* aro = &at[col][0];   // A row m=col (LDS)
    for (int dt = 0; dt < D_DIM / 16; ++dt) {
        const unsigned short* bcolp = vtT + ((size_t)b * D_DIM + dt * 16 + col) * L_PAD; // B col n=col -> d
        v8f c; for (int r = 0; r < 8; ++r) c[r] = 0.f;
        for (int kk = 0; kk < NLT * 16 / 32; ++kk) {  // 7 k-steps over 224 tokens
            const int ka = kk * 32 + 8 * hi;
            uint4 a0 = *(const uint4*)(aro + ka);
            uint4 a1 = *(const uint4*)(aro + ka + 16);
            const int kb = kk * 32 + 8 * hi;
            uint4 b0 = *(const uint4*)(bcolp + kb);
            uint4 b1 = *(const uint4*)(bcolp + kb + 16);
            c = __builtin_amdgcn_wmma_f32_16x16x32_bf16(
                    false, mk16(a0, a1), false, mk16(b0, b1), (short)0, c, false, false);
        }
        for (int r = 0; r < 8; ++r) {
            float v = c[r];                                   // aggregated[m = r+8*hi][d = dt*16+col]
            accS[r] += v * v;
            accD[r] += v * qbase[(size_t)(r + 8 * hi) * D_DIM + dt * 16 + col];
        }
    }
    // reduce across the 16 lanes of each half-wave (rows r+8*hi)
    for (int r = 0; r < 8; ++r) {
        float s = accS[r], d = accD[r];
        for (int m = 1; m < 16; m <<= 1) {
            s += __shfl_xor(s, m, 32);
            d += __shfl_xor(d, m, 32);
        }
        if (col == 0) {
            const int n = ntile * 16 + r + 8 * hi;
            logits[(size_t)n * B_IMG + b] = d / fmaxf(sqrtf(s), 1e-12f);
        }
    }
}

// ================= Kernel D1: per-column sums of exp(logits/temp) =================
__global__ __launch_bounds__(256)
void colsum_kernel(const float* __restrict__ logits, const int* __restrict__ gmap,
                   const float* __restrict__ logT,
                   float* __restrict__ colsum, float* __restrict__ colpos) {
    __shared__ float b1[256], b2[256];
    const int b = blockIdx.x, tid = threadIdx.x;
    const float scale = expf(-logT[0]);   // 1/temp
    float cs = 0.f, cp = 0.f;
    for (int n = tid; n < N_TXT; n += 256) {
        float e = expf(logits[(size_t)n * B_IMG + b] * scale);
        cs += e;
        if (gmap[n] == b) cp += e;
    }
    const float CS = block_reduce(cs, b1, tid, 256);
    const float CP = block_reduce(cp, b2, tid, 256);
    if (tid == 0) { colsum[b] = CS; colpos[b] = CP; }
}

// ================= Kernel D2: final NCE loss =================
__global__ __launch_bounds__(1024)
void loss_kernel(const float* __restrict__ logits, const int* __restrict__ gmap,
                 const float* __restrict__ logT,
                 const float* __restrict__ colsum, const float* __restrict__ colpos,
                 float* __restrict__ out) {
    __shared__ float buf[1024];
    const int n = threadIdx.x;
    const float scale = expf(-logT[0]);
    float rowsum = 0.f;
    for (int b = 0; b < B_IMG; ++b)
        rowsum += expf(logits[(size_t)n * B_IMG + b] * scale);
    const int g = gmap[n];
    const float pos = expf(logits[(size_t)n * B_IMG + g] * scale);
    const float row_loss = -logf(pos / (rowsum + 1e-8f) + 1e-8f);
    const float sum_neg  = colsum[g] - colpos[g];
    const float col_loss = -logf(pos / (pos + sum_neg + 1e-8f) + 1e-8f);
    const float total = block_reduce(row_loss + col_loss, buf, n, 1024);
    if (n == 0) out[0] = total / (2.f * N_TXT);
}

// ================= launch =================
extern "C" void kernel_launch(void* const* d_in, const int* in_sizes, int n_in,
                              void* d_out, int out_size, void* d_ws, size_t ws_size,
                              hipStream_t stream) {
    const float* text   = (const float*)d_in[0];
    const float* vision = (const float*)d_in[1];
    const int*   gmap   = (const int*)d_in[2];
    const float* gamma  = (const float*)d_in[3];
    const float* beta   = (const float*)d_in[4];
    const float* logT   = (const float*)d_in[5];
    float* out = (float*)d_out;

    char* w = (char*)d_ws;
    unsigned short* tf16 = (unsigned short*)(w + OFF_TF16);
    float*          qn   = (float*)(w + OFF_QN);
    unsigned short* vt16 = (unsigned short*)(w + OFF_VT16);
    unsigned short* vtT  = (unsigned short*)(w + OFF_VTT);
    float*          lg   = (float*)(w + OFF_LOG);
    float*          cs   = (float*)(w + OFF_CS);
    float*          cp   = (float*)(w + OFF_CP);

    ln_text_kernel<<<N_TXT, 256, 0, stream>>>(text, gamma, beta, tf16, qn);
    ln_vis_kernel<<<dim3(L_PAD, B_IMG), 256, 0, stream>>>(vision, gamma, beta, vt16, vtT);
    attn_kernel<<<dim3(N_TXT / 16, B_IMG), 32, 0, stream>>>(tf16, qn, vt16, vtT, lg);
    colsum_kernel<<<B_IMG, 256, 0, stream>>>(lg, gmap, logT, cs, cp);
    loss_kernel<<<1, 1024, 0, stream>>>(lg, gmap, logT, cs, cp, out);
    (void)in_sizes; (void)n_in; (void)out_size; (void)ws_size;
}
